// KernelNetwork_71116068488013
// MI455X (gfx1250) — compile-verified, matrix operand order checked by reading, not run on hardware
//
#include <hip/hip_runtime.h>
#include <stdint.h>

// ---------------- problem constants ----------------
#define PK_N   (512 * 512)      // number of cells
#define DYNC   8
#define HID    64
#define D_IN   136              // 8 + 8*16
#define KP1    160              // K of layer-1 GEMM padded to multiple of 32
#define NPOSTT 9                // ceil(136/16) N-tiles for post layer
#define WAVES  4                // waves (16-cell tiles) per block

// workspace/LDS weight-fragment layout (dwords)
#define WS_PRE   0              // 5 KC * 4 NT * 256  = 5120
#define WS_IH    5120           // 2 * 16 * 256       = 8192
#define WS_HH    13312          // 2 * 16 * 256       = 8192
#define WS_POST  21504          // 2 *  9 * 256       = 4608
#define WS_TOTAL 26112          // dwords = 104448 bytes

typedef __attribute__((ext_vector_type(16))) _Float16 v16h;
typedef __attribute__((ext_vector_type(8)))  float    v8f;

union FragU { v16h h; uint4 q[2]; };

// ---- fast activations ----------------------------------------------------
__device__ inline float sigm(float x) {
  float e = __builtin_amdgcn_exp2f(x * -1.44269504f);
  return __builtin_amdgcn_rcpf(1.0f + e);
}

#if __has_builtin(__builtin_amdgcn_tanhf)
__device__ inline float ftanh(float x) { return __builtin_amdgcn_tanhf(x); }
#else
__device__ inline float ftanh(float x) {  // 2*sigmoid(2x) - 1
  float e = __builtin_amdgcn_exp2f(x * -2.88539008f);
  return __builtin_fmaf(2.0f, __builtin_amdgcn_rcpf(1.0f + e), -1.0f);
}
#endif

// ---- helpers -------------------------------------------------------------

__device__ inline uint32_t pack2h(float a, float b) {
  union { _Float16 h[2]; uint32_t u; } x;
  x.h[0] = (_Float16)a; x.h[1] = (_Float16)b;
  return x.u;
}

// A-operand fragment (16x32 f16) from a row-major f16 LDS tile.
__device__ inline v16h load_a_lds(const _Float16* tile, int pitch, int kc, int lane) {
  const int m   = lane & 15;
  const int sel = lane >> 4;
  const uint4* p = (const uint4*)(tile + m * pitch + kc * 32 + sel * 8);
  FragU f;
  f.q[0] = p[0];      // K = kc*32 + sel*8 .. +7
  f.q[1] = p[2];      // K = kc*32 + 16 + sel*8 .. +7
  return f.h;
}

// B-operand fragment from LDS-resident pre-swizzled weights: dword [lane*8+v].
__device__ inline v16h load_b_frag(const uint32_t* frag, int lane) {
  const uint4* p = (const uint4*)frag + lane * 2;
  FragU f;
  f.q[0] = p[0];
  f.q[1] = p[1];
  return f.h;
}

__device__ inline v8f wmma_f16(v16h a, v16h b, v8f c) {
  return __builtin_amdgcn_wmma_f32_16x16x32_f16(false, a, false, b, (short)0, c,
                                                false, false);
}

// ---- prep: re-pack an f32 KxN weight matrix into WMMA B-fragment order ---
__global__ void pack_b_kernel(const float* __restrict__ W, int K, int Ncols,
                              int KC, int NT, uint32_t* __restrict__ out) {
  int t = blockIdx.x * blockDim.x + threadIdx.x;
  int total = KC * NT * 256;
  if (t >= total) return;
  int v    = t & 7;
  int lane = (t >> 3) & 31;
  int frag = t >> 8;
  int kc = frag % KC;
  int nt = frag / KC;
  int col = nt * 16 + (lane & 15);
  int k0  = ((v & 4) ? 16 : 0) + ((lane >> 4) ? 8 : 0) + (v & 3) * 2;
  int k   = kc * 32 + k0;
  float a0 = (k     < K && col < Ncols) ? W[(size_t)k       * Ncols + col] : 0.0f;
  float a1 = (k + 1 < K && col < Ncols) ? W[(size_t)(k + 1) * Ncols + col] : 0.0f;
  out[(size_t)frag * 256 + lane * 8 + v] = pack2h(a0, a1);
}

// ---- one gate group (64 gate cols = 4 N-tiles): acc = pre@Wih + h@Whh + b -
__device__ inline void gate_group(v8f acc[4], int g,
                                  const _Float16* spre, const _Float16* sh,
                                  const uint32_t* fr_ih, const uint32_t* fr_hh,
                                  const float* b_lstm, int lane, int n) {
  const v8f zero = {0, 0, 0, 0, 0, 0, 0, 0};
#pragma unroll
  for (int t = 0; t < 4; ++t) acc[t] = zero;
#pragma unroll
  for (int kc = 0; kc < 2; ++kc) {
    v16h ap = load_a_lds(spre, HID, kc, lane);
    v16h ah = load_a_lds(sh,   HID, kc, lane);
#pragma unroll
    for (int t = 0; t < 4; ++t) {
      int nt = g * 4 + t;
      acc[t] = wmma_f16(ap, load_b_frag(fr_ih + (nt * 2 + kc) * 256, lane), acc[t]);
      acc[t] = wmma_f16(ah, load_b_frag(fr_hh + (nt * 2 + kc) * 256, lane), acc[t]);
    }
  }
#pragma unroll
  for (int t = 0; t < 4; ++t) {
    float bb = b_lstm[g * 64 + t * 16 + n];
#pragma unroll
    for (int v = 0; v < 8; ++v) acc[t][v] += bb;
  }
}

// ---- fused network step ---------------------------------------------------
__global__ void __launch_bounds__(WAVES * 32)
knet_fused(const float* __restrict__ dyn_in, const float* __restrict__ lat_prev,
           const float* __restrict__ c_in,  const float* __restrict__ h_in,
           const int*   __restrict__ coming_from,
           const uint32_t* __restrict__ wsrc,   // all weight fragments (dwords)
           const float* __restrict__ b_pre, const float* __restrict__ b_lstm,
           const float* __restrict__ b_post,
           float* __restrict__ out_dyn, float* __restrict__ out_lat,
           float* __restrict__ out_c,   float* __restrict__ out_h) {
  // Block-shared weight fragments (102 KB) + per-wave staging tiles (45 KB):
  // 147 KB of the 320 KB/WGP LDS -> 2 blocks per WGP.
  __shared__ __align__(16) uint32_t s_w[WS_TOTAL];
  __shared__ __align__(16) _Float16 s_x  [WAVES][16 * KP1];
  __shared__ __align__(16) _Float16 s_pre[WAVES][16 * HID];
  __shared__ __align__(16) _Float16 s_h  [WAVES][16 * HID];
  __shared__ __align__(16) _Float16 s_h2 [WAVES][16 * HID];

  const int lane = threadIdx.x & 31;
  const int wave = threadIdx.x >> 5;
  const size_t cellbase = ((size_t)blockIdx.x * WAVES + wave) * 16;

  // ---- cooperative weight stage: 6528 uint4 over 128 threads, 3-deep ----
  {
    const uint4* src = (const uint4*)wsrc;
    uint4* dst = (uint4*)s_w;
    for (int i = threadIdx.x; i < WS_TOTAL / 4; i += 3 * WAVES * 32) {
      uint4 a0 = src[i];
      uint4 a1 = src[i + 128];
      uint4 a2 = src[i + 256];
      dst[i]       = a0;
      dst[i + 128] = a1;
      dst[i + 256] = a2;
    }
  }

  _Float16* sx   = s_x[wave];
  _Float16* spre = s_pre[wave];
  _Float16* sh   = s_h[wave];
  _Float16* sh2  = s_h2[wave];

  // ---- stage x-tile (16 x 160 f16) into LDS ----
  {  // dyn part: 128 contiguous floats
    const float4* dp = (const float4*)(dyn_in + cellbase * DYNC);
    float4 d4 = dp[lane];
    int m = lane >> 1, col = (lane & 1) * 4;
    *(uint2*)(sx + m * KP1 + col) =
        make_uint2(pack2h(d4.x, d4.y), pack2h(d4.z, d4.w));
  }
#pragma unroll
  for (int it = 0; it < 4; ++it) {  // lat gather: 16 cells x 8 dirs
    int p = it * 32 + lane;
    int m = p >> 3, d = p & 7;
    int src = coming_from[(size_t)d * PK_N + cellbase + m];
    const float4* q = (const float4*)(lat_prev + (size_t)src * 128 + d * 16);
    float4 q0 = q[0], q1 = q[1], q2 = q[2], q3 = q[3];
    _Float16* dst = sx + m * KP1 + DYNC + d * 16;
    *(uint4*)dst = make_uint4(pack2h(q0.x, q0.y), pack2h(q0.z, q0.w),
                              pack2h(q1.x, q1.y), pack2h(q1.z, q1.w));
    *(uint4*)(dst + 8) = make_uint4(pack2h(q2.x, q2.y), pack2h(q2.z, q2.w),
                                    pack2h(q3.x, q3.y), pack2h(q3.z, q3.w));
  }
  for (int idx = lane; idx < 16 * 12; idx += 32) {  // zero K-pad cols 136..159
    int m = idx / 12, j = idx % 12;
    *(uint32_t*)(sx + m * KP1 + D_IN + j * 2) = 0u;
  }
  {  // stage previous h (16 x 64) as f16
    const float4* hp = (const float4*)(h_in + cellbase * HID);
#pragma unroll
    for (int i = lane; i < 256; i += 32) {
      float4 hv = hp[i];
      *(uint2*)(sh + i * 4) = make_uint2(pack2h(hv.x, hv.y), pack2h(hv.z, hv.w));
    }
  }

  __syncthreads();  // weights visible to all waves

  const uint32_t* fr_pre  = s_w + WS_PRE;
  const uint32_t* fr_ih   = s_w + WS_IH;
  const uint32_t* fr_hh   = s_w + WS_HH;
  const uint32_t* fr_post = s_w + WS_POST;

  const int n      = lane & 15;         // C/D column within tile
  const int rowoff = (lane >> 4) * 8;   // C/D row offset for hi lanes
  const v8f zero8  = {0, 0, 0, 0, 0, 0, 0, 0};

  // ---- layer 1: pre = tanh(x @ W_pre + b_pre), 5 K-chunks x 4 N-tiles ----
  {
    v8f acc[4] = {zero8, zero8, zero8, zero8};
#pragma unroll
    for (int kc = 0; kc < 5; ++kc) {
      v16h a = load_a_lds(sx, KP1, kc, lane);
#pragma unroll
      for (int t = 0; t < 4; ++t)
        acc[t] = wmma_f16(a, load_b_frag(fr_pre + (t * 5 + kc) * 256, lane), acc[t]);
    }
#pragma unroll
    for (int t = 0; t < 4; ++t) {
      float bb = b_pre[t * 16 + n];
#pragma unroll
      for (int v = 0; v < 8; ++v)
        spre[(v + rowoff) * HID + t * 16 + n] = (_Float16)ftanh(acc[t][v] + bb);
    }
  }

  // ---- LSTM gates: i (g=0), g (g=2), f (g=1), o (g=3) ----
  v8f gi[4], tmp[4], cnew[4];
  gate_group(gi, 0, spre, sh, fr_ih, fr_hh, b_lstm, lane, n);   // input gate
  gate_group(tmp, 2, spre, sh, fr_ih, fr_hh, b_lstm, lane, n);  // candidate g
#pragma unroll
  for (int t = 0; t < 4; ++t)
#pragma unroll
    for (int v = 0; v < 8; ++v)
      gi[t][v] = sigm(gi[t][v]) * ftanh(tmp[t][v]);             // i*g

  gate_group(tmp, 1, spre, sh, fr_ih, fr_hh, b_lstm, lane, n);  // forget gate
#pragma unroll
  for (int t = 0; t < 4; ++t)
#pragma unroll
    for (int v = 0; v < 8; ++v) {
      size_t cell = cellbase + rowoff + v;
      float cold = c_in[cell * HID + t * 16 + n];
      float cv   = sigm(tmp[t][v]) * cold + gi[t][v];
      cnew[t][v] = cv;
      out_c[cell * HID + t * 16 + n] = cv;
    }

  gate_group(tmp, 3, spre, sh, fr_ih, fr_hh, b_lstm, lane, n);  // output gate
#pragma unroll
  for (int t = 0; t < 4; ++t)
#pragma unroll
    for (int v = 0; v < 8; ++v) {
      size_t cell = cellbase + rowoff + v;
      float hv = sigm(tmp[t][v]) * ftanh(cnew[t][v]);
      out_h[cell * HID + t * 16 + n] = hv;
      sh2[(v + rowoff) * HID + t * 16 + n] = (_Float16)hv;
    }

  // ---- post layer: post = tanh(h' @ W_post + b_post); split dyn/lat ----
  {
    v16h a0 = load_a_lds(sh2, HID, 0, lane);
    v16h a1 = load_a_lds(sh2, HID, 1, lane);
#pragma unroll
    for (int nt = 0; nt < NPOSTT; ++nt) {
      v8f acc = zero8;
      acc = wmma_f16(a0, load_b_frag(fr_post + (nt * 2 + 0) * 256, lane), acc);
      acc = wmma_f16(a1, load_b_frag(fr_post + (nt * 2 + 1) * 256, lane), acc);
      int col = nt * 16 + n;
      if (col < D_IN) {
        float bb = b_post[col];
#pragma unroll
        for (int v = 0; v < 8; ++v) {
          float val = ftanh(acc[v] + bb);
          size_t cell = cellbase + rowoff + v;
          if (col < DYNC) out_dyn[cell * DYNC + col] = val;
          else            out_lat[cell * 128 + (col - DYNC)] = val;
        }
      }
    }
  }
}

extern "C" void kernel_launch(void* const* d_in, const int* in_sizes, int n_in,
                              void* d_out, int out_size, void* d_ws, size_t ws_size,
                              hipStream_t stream) {
  (void)in_sizes; (void)n_in; (void)out_size; (void)ws_size;

  const float* dyn_in      = (const float*)d_in[0];
  const float* lat_prev    = (const float*)d_in[1];
  const float* c_in        = (const float*)d_in[2];
  const float* h_in        = (const float*)d_in[3];
  const int*   coming_from = (const int*)  d_in[5];
  const float* W_pre       = (const float*)d_in[7];
  const float* b_pre       = (const float*)d_in[8];
  const float* W_ih        = (const float*)d_in[9];
  const float* W_hh        = (const float*)d_in[10];
  const float* b_lstm      = (const float*)d_in[11];
  const float* W_post      = (const float*)d_in[12];
  const float* b_post      = (const float*)d_in[13];

  float* out     = (float*)d_out;
  float* out_dyn = out;
  float* out_lat = out + (size_t)PK_N * 8;
  float* out_c   = out + (size_t)PK_N * 136;
  float* out_h   = out + (size_t)PK_N * 200;

  uint32_t* ws = (uint32_t*)d_ws;

  // Re-pack weights into WMMA B-fragment layout (L2-resident, ~100 KB).
  pack_b_kernel<<<(5 * 4  * 256 + 255) / 256, 256, 0, stream>>>(W_pre, D_IN, HID, 5, 4,  ws + WS_PRE);
  pack_b_kernel<<<(2 * 16 * 256 + 255) / 256, 256, 0, stream>>>(W_ih,  HID, 256, 2, 16, ws + WS_IH);
  pack_b_kernel<<<(2 * 16 * 256 + 255) / 256, 256, 0, stream>>>(W_hh,  HID, 256, 2, 16, ws + WS_HH);
  pack_b_kernel<<<(2 * 9  * 256 + 255) / 256, 256, 0, stream>>>(W_post, HID, D_IN, 2, 9, ws + WS_POST);

  // Fused step: 16384 16-cell tiles, one per wave32, WAVES waves per block.
  const int tiles  = PK_N / 16;
  const int blocks = tiles / WAVES;
  knet_fused<<<blocks, WAVES * 32, 0, stream>>>(
      dyn_in, lat_prev, c_in, h_in, coming_from, ws,
      b_pre, b_lstm, b_post,
      out_dyn, out_lat, out_c, out_h);
}
